// ImprovedPCENLayer_58102317580724
// MI455X (gfx1250) — matrix operand order repaired and checked by reading, not compiled, not run on hardware
//
#include <hip/hip_runtime.h>

#define FLOORV 1e-6f
#define RPB  128   // output rows per block
#define XSTR 132   // LDS float stride for fp32 X tile
#define HSTR 136   // LDS ushort stride for bf16 tiles (mult of 8 -> 16B aligned rows)

typedef __attribute__((ext_vector_type(16))) __bf16 v16bf;
typedef __attribute__((ext_vector_type(8)))  float  v8f;

static __device__ __forceinline__ unsigned short f2bf(float f) {
    // round-to-nearest-even float -> bf16 bits (inputs are finite positive)
    unsigned int u = __float_as_uint(f);
    u += 0x7FFFu + ((u >> 16) & 1u);
    return (unsigned short)(u >> 16);
}

// pow for strictly-positive x via native v_log_f32 / v_exp_f32 (log2/exp2).
static __device__ __forceinline__ float fast_pow(float x, float y) {
    return __builtin_amdgcn_exp2f(y * __builtin_amdgcn_logf(x));
}

// Build B[j][w] = alpha^(511 + j - w) / sum_{i<1024} alpha^i, stored bf16,
// row-major with w as the slow index: Bg[w*128 + j].
__global__ __launch_bounds__(256)
void pcen_build_B(const float* __restrict__ alpha_p, unsigned short* __restrict__ Bg) {
    int i = blockIdx.x * 256 + threadIdx.x;
    if (i >= 128 * 128) return;
    int w = i >> 7, j = i & 127;
    float a = alpha_p[0];
    float S = (fabsf(1.0f - a) > 1e-7f)
                  ? (1.0f - fast_pow(a, 1024.0f)) / (1.0f - a)
                  : 1024.0f;
    float kv = fast_pow(a, (float)(511 + j - w)) / S;
    Bg[i] = f2bf(kv);
}

// Fused PCEN epilogue for one 16x16 C tile (8 f32 acc regs per lane).
template <bool GUARD>
static __device__ __forceinline__ void pcen_epilogue(
    const v8f& acc, int rt, int half, int ocol, long rowBase, int totalRows,
    const float* Xs, float* out, float nalpha, float delta, float root, float droot) {
#pragma unroll
    for (int r = 0; r < 8; ++r) {
        int  orow = rt * 16 + half * 8 + r;
        long gr   = rowBase + orow;
        if (!GUARD || gr < totalRows) {
            float s     = fmaxf(acc[r], FLOORV);
            float xv    = Xs[orow * XSTR + ocol];
            float inner = xv * fast_pow(s, nalpha) + delta;
            out[gr * 128 + ocol] = fast_pow(inner, root) - droot;
        }
    }
}

__global__ __launch_bounds__(256)
void pcen_wmma_kernel(const float* __restrict__ x,
                      const float* __restrict__ alpha_p,
                      const float* __restrict__ delta_p,
                      const float* __restrict__ root_p,
                      const unsigned short* __restrict__ Bg,  // may be null
                      float* __restrict__ out,
                      int totalRows) {
    __shared__ __align__(16) float          Xs[RPB * XSTR];   // fp32 X (epilogue)
    __shared__ __align__(16) unsigned short Xb[RPB * HSTR];   // bf16 X (WMMA A)
    __shared__ __align__(16) unsigned short Bs[128 * HSTR];   // bf16 B (WMMA B)

    const int tid  = threadIdx.x;
    const int wave = tid >> 5;      // 8 waves, wave32
    const int lane = tid & 31;
    const int half = lane >> 4;     // 0: lanes 0-15, 1: lanes 16-31
    const int mrow = lane & 15;

    const float alpha = alpha_p[0];
    const float delta = delta_p[0];
    const float root  = root_p[0];

    const long rowBase  = (long)blockIdx.x * RPB;
    const bool fullTile = (rowBase + RPB) <= (long)totalRows;

    // ---- Stage X tile: floored fp32 copy + bf16 copy, one pass ----
    for (int i = tid; i < RPB * 32; i += 256) {
        int  r  = i >> 5;
        int  c4 = i & 31;
        long gr = rowBase + r;
        if (gr >= totalRows) gr = totalRows - 1;   // clamp (still valid memory)
        float4 v = *((const float4*)(x + gr * 128) + c4);
        v.x = fmaxf(v.x, FLOORV);
        v.y = fmaxf(v.y, FLOORV);
        v.z = fmaxf(v.z, FLOORV);
        v.w = fmaxf(v.w, FLOORV);
        *(float4*)&Xs[r * XSTR + c4 * 4] = v;
        unsigned int p0 = ((unsigned int)f2bf(v.y) << 16) | (unsigned int)f2bf(v.x);
        unsigned int p1 = ((unsigned int)f2bf(v.w) << 16) | (unsigned int)f2bf(v.z);
        uint2 pk; pk.x = p0; pk.y = p1;
        *(uint2*)&Xb[r * HSTR + c4 * 4] = pk;
    }

    // ---- Stage B (128x128 bf16) into LDS ----
    if (Bg) {
        const unsigned int* Bg32 = (const unsigned int*)Bg;
        for (int i = tid; i < 128 * 64; i += 256) {
            int r = i >> 6, c = i & 63;
            *(unsigned int*)&Bs[r * HSTR + c * 2] = Bg32[r * 64 + c];
        }
    } else {
        // fallback: compute B locally (only if workspace too small)
        float S = (fabsf(1.0f - alpha) > 1e-7f)
                      ? (1.0f - fast_pow(alpha, 1024.0f)) / (1.0f - alpha)
                      : 1024.0f;
        float invS = 1.0f / S;
        for (int i = tid; i < 128 * 128; i += 256) {
            int w = i >> 7, j = i & 127;
            Bs[w * HSTR + j] = f2bf(fast_pow(alpha, (float)(511 + j - w)) * invS);
        }
    }
    __syncthreads();

    // ---- Per-wave B fragments: N-tile = wave, all 4 K-chunks (K=128) ----
    // B (32x16 K-major): lanes 0-15 hold K {kb..kb+7, kb+16..kb+23} of column mrow,
    // lanes 16-31 hold K {kb+8..kb+15, kb+24..kb+31}.
    const int col0 = wave * 16;
    v16bf bfrag[4];
#pragma unroll
    for (int kc = 0; kc < 4; ++kc) {
        const unsigned short* bp = &Bs[(col0 + mrow) * HSTR + kc * 32 + half * 8];
        union { v16bf v; uint4 q[2]; } t;
        t.q[0] = *(const uint4*)(bp);
        t.q[1] = *(const uint4*)(bp + 16);
        bfrag[kc] = t.v;
    }

    const float nalpha = -alpha;
    const float droot  = fast_pow(delta, root);
    const int   ocol   = col0 + mrow;

    // ---- 8 row tiles per wave, processed 2 at a time for WMMA overlap ----
#pragma unroll
    for (int rt2 = 0; rt2 < RPB / 32; ++rt2) {
        const int rtA = rt2 * 2;
        const int rtB = rt2 * 2 + 1;
        v8f acc0 = {0.f, 0.f, 0.f, 0.f, 0.f, 0.f, 0.f, 0.f};
        v8f acc1 = {0.f, 0.f, 0.f, 0.f, 0.f, 0.f, 0.f, 0.f};
        const unsigned short* xrow0 = &Xb[(rtA * 16 + mrow) * HSTR];
        const unsigned short* xrow1 = &Xb[(rtB * 16 + mrow) * HSTR];
#pragma unroll
        for (int kc = 0; kc < 4; ++kc) {
            // A (16x32): lane<16 -> K {kb..kb+7, kb+16..kb+23},
            //            lane>=16 -> K {kb+8..kb+15, kb+24..kb+31}
            const int s0 = kc * 32 + half * 8;
            const int s1 = s0 + 16;
            union { v16bf v; uint4 q[2]; } a0, a1;
            a0.q[0] = *(const uint4*)(xrow0 + s0);
            a0.q[1] = *(const uint4*)(xrow0 + s1);
            a1.q[0] = *(const uint4*)(xrow1 + s0);
            a1.q[1] = *(const uint4*)(xrow1 + s1);
            acc0 = __builtin_amdgcn_wmma_f32_16x16x32_bf16(
                false, a0.v, false, bfrag[kc], (short)0, acc0, false, false);
            acc1 = __builtin_amdgcn_wmma_f32_16x16x32_bf16(
                false, a1.v, false, bfrag[kc], (short)0, acc1, false, false);
        }
        // C layout: VGPR r -> row rt*16 + half*8 + r, col = col0 + mrow
        if (fullTile) {
            pcen_epilogue<false>(acc0, rtA, half, ocol, rowBase, totalRows,
                                 Xs, out, nalpha, delta, root, droot);
            pcen_epilogue<false>(acc1, rtB, half, ocol, rowBase, totalRows,
                                 Xs, out, nalpha, delta, root, droot);
        } else {
            pcen_epilogue<true>(acc0, rtA, half, ocol, rowBase, totalRows,
                                Xs, out, nalpha, delta, root, droot);
            pcen_epilogue<true>(acc1, rtB, half, ocol, rowBase, totalRows,
                                Xs, out, nalpha, delta, root, droot);
        }
    }
}

extern "C" void kernel_launch(void* const* d_in, const int* in_sizes, int n_in,
                              void* d_out, int out_size, void* d_ws, size_t ws_size,
                              hipStream_t stream) {
    const float* x       = (const float*)d_in[0];
    const float* alpha_p = (const float*)d_in[1];
    const float* delta_p = (const float*)d_in[2];
    const float* root_p  = (const float*)d_in[3];
    float*       out     = (float*)d_out;

    const int totalRows = in_sizes[0] / 128;   // 64*1024 = 65536 rows of 128 mels

    unsigned short* Bg = nullptr;
    if (ws_size >= (size_t)(128 * 128 * sizeof(unsigned short))) {
        Bg = (unsigned short*)d_ws;
        pcen_build_B<<<(128 * 128 + 255) / 256, 256, 0, stream>>>(alpha_p, Bg);
    }

    const int blocks = (totalRows + RPB - 1) / RPB;
    pcen_wmma_kernel<<<blocks, 256, 0, stream>>>(x, alpha_p, delta_p, root_p,
                                                 Bg, out, totalRows);
}
